// MiniFastSpeech_83030307766312
// MI455X (gfx1250) — compile-verified, matrix operand order checked by reading, not run on hardware
//
#include <hip/hip_runtime.h>
#include <hip/hip_bf16.h>

#define VOCAB 256
#define EMB   128
#define HID   256
#define MEL   80
#define BB    64
#define TT    512
#define GH    1024          // 4*HID
#define KA    384           // HID + EMB (recurrent K dim)
#define AS    392           // LDS A-buffer stride in f16 (padded vs 384)

typedef __attribute__((ext_vector_type(16))) _Float16 v16h;
typedef __attribute__((ext_vector_type(8)))  _Float16 v8h;
typedef __attribute__((ext_vector_type(4)))  _Float16 v4h;
typedef __attribute__((ext_vector_type(8)))  float    v8f;

// K index inside a 16-bit WMMA A/B fragment for (lane, element i) per ISA layout:
// lane<16 holds K {0..7,16..23}; lane>=16 holds K {8..15,24..31}.
__device__ __forceinline__ int frag_k(int lane, int i) {
    return ((i < 8) ? i : (i + 8)) + ((lane & 16) ? 8 : 0);
}

__device__ __forceinline__ v8f v8f_zero() {
    v8f z;
#pragma unroll
    for (int i = 0; i < 8; ++i) z[i] = 0.0f;
    return z;
}

__device__ __forceinline__ v16h combine16(v8h lo, v8h hi) {
    v16h r;
#pragma unroll
    for (int i = 0; i < 8; ++i) { r[i] = lo[i]; r[i + 8] = hi[i]; }
    return r;
}

// ---- branch-free activations on native TRANS ops (co-execute with matrix pipe) ----
#define LOG2E 1.4426950408889634f

#if __has_builtin(__builtin_amdgcn_tanhf)
__device__ __forceinline__ float fast_tanh(float x) {
    return __builtin_amdgcn_tanhf(x);          // gfx1250 V_TANH_F32 (confirmed in asm)
}
__device__ __forceinline__ float fast_sigmoid(float x) {
    // 1 TRANS + 2 VALU instead of 2 TRANS + 2 VALU
    return __builtin_fmaf(0.5f, fast_tanh(0.5f * x), 0.5f);
}
#else
__device__ __forceinline__ float fast_tanh(float x) {
    float e = __builtin_amdgcn_exp2f(2.0f * LOG2E * x);   // exp(2x), full-range HW exp2
    return (e - 1.0f) * __builtin_amdgcn_rcpf(e + 1.0f);
}
__device__ __forceinline__ float fast_sigmoid(float x) {
    return __builtin_amdgcn_rcpf(1.0f + __builtin_amdgcn_exp2f(-LOG2E * x));
}
#endif

__device__ __forceinline__ v8f wmma_f16(v16h a, v16h b, v8f c) {
    return __builtin_amdgcn_wmma_f32_16x16x32_f16(false, a, false, b, (short)0, c,
                                                  false, false);
}

// ---------------- prep kernels ----------------

__global__ void k_prep_embed(const float* __restrict__ e, _Float16* __restrict__ e16) {
    int i = blockIdx.x * blockDim.x + threadIdx.x;
    if (i < VOCAB * EMB) e16[i] = (_Float16)e[i];
}

__global__ void k_prep_bias(const float* __restrict__ bih, const float* __restrict__ bhh,
                            float* __restrict__ bias) {
    int i = blockIdx.x * blockDim.x + threadIdx.x;
    if (i < GH) bias[i] = bih[i] + bhh[i];
}

// Pack W = [whh (K 0..255) | wih (K 256..383)]  (GH x KA) into WMMA B-fragment tiles.
// Tile index t = nt*12 + kt  (nt over N=1024/16, kt over K=384/32); 512 f16 per tile,
// per-lane 16 contiguous f16.
__global__ void k_pack_gates(const float* __restrict__ whh, const float* __restrict__ wih,
                             _Float16* __restrict__ dst) {
    int idx = blockIdx.x * blockDim.x + threadIdx.x;
    if (idx >= 64 * 12 * 512) return;
    int within = idx & 511;
    int t      = idx >> 9;
    int lane   = within >> 4;
    int i      = within & 15;
    int nt = t / 12, kt = t % 12;
    int n = nt * 16 + (lane & 15);
    int k = kt * 32 + frag_k(lane, i);
    float v = (k < HID) ? whh[n * HID + k] : wih[n * EMB + (k - HID)];
    dst[idx] = (_Float16)v;
}

// Pack lin_w (MEL x 2*HID) into B-fragment tiles: t = nt*16 + kt (nt over 80/16, kt over 512/32).
__global__ void k_pack_lin(const float* __restrict__ lw, _Float16* __restrict__ dst) {
    int idx = blockIdx.x * blockDim.x + threadIdx.x;
    if (idx >= 5 * 16 * 512) return;
    int within = idx & 511;
    int t      = idx >> 9;
    int lane   = within >> 4;
    int i      = within & 15;
    int nt = t >> 4, kt = t & 15;
    int n = nt * 16 + (lane & 15);
    int k = kt * 32 + frag_k(lane, i);
    dst[idx] = (_Float16)lw[n * (2 * HID) + k];
}

// ---------------- duration + cumsum ----------------

__global__ void k_dur_cum(const int* __restrict__ x, const float* __restrict__ embed,
                          const float* __restrict__ dp_w, const float* __restrict__ dp_b,
                          int* __restrict__ cum) {
    __shared__ int s[TT];
    int b = blockIdx.x, t = threadIdx.x;          // blockDim = 512
    int v = x[b * TT + t];
    const float* er = embed + v * EMB;
    float acc = dp_b[0];
#pragma unroll 8
    for (int e = 0; e < EMB; ++e) acc += er[e] * dp_w[e];
    float d = acc > 0.0f ? acc : 0.0f;
    s[t] = (int)d + 1;                            // floor(relu)+1
    __syncthreads();
    for (int off = 1; off < TT; off <<= 1) {      // inclusive Hillis-Steele scan
        int val = (t >= off) ? s[t - off] : 0;
        __syncthreads();
        s[t] += val;
        __syncthreads();
    }
    cum[b * TT + t] = s[t];
}

// ---------------- length-regulator expansion ----------------

__global__ void k_expand(const int* __restrict__ x, const int* __restrict__ cum,
                         const _Float16* __restrict__ e16, _Float16* __restrict__ exp16,
                         int L) {
    int b    = blockIdx.y;
    int wave = threadIdx.x >> 5;
    int lane = threadIdx.x & 31;
    int pos  = blockIdx.x * 8 + wave;
    if (pos >= L) return;
    const int* crow = cum + b * TT;
    int len = crow[TT - 1];
    v4h val;
#pragma unroll
    for (int i = 0; i < 4; ++i) val[i] = (_Float16)0.0f;
    if (pos < len) {
        int lo = 0, hi = TT;                      // searchsorted(side='right')
        while (lo < hi) {
            int mid = (lo + hi) >> 1;
            if (crow[mid] <= pos) lo = mid + 1; else hi = mid;
        }
        int idx = lo > TT - 1 ? TT - 1 : lo;
        int vv  = x[b * TT + idx];
        val = *(const v4h*)(e16 + vv * EMB + lane * 4);
    }
    *(v4h*)(exp16 + ((size_t)(b * L + pos)) * EMB + lane * 4) = val;
}

// ---------------- bidirectional LSTM recurrence ----------------
// grid = (4 batch-tiles, 2 directions); block = 512 (16 waves, 4 per SIMD32).
// LDS a_buf holds A = [h (K 0..255) | x_t (K 256..383)] for 16 batch rows.
// Wave w owns hidden-unit tile ht = w; its gate tiles are nt = {0,16,32,48}+ht.
// Next step's x_t is double-buffered through registers so its global latency
// hides under the 48-WMMA compute phase.

__global__ __launch_bounds__(512, 1) void k_lstm(
    const _Float16* __restrict__ exp16,
    const _Float16* __restrict__ wpack_f, const _Float16* __restrict__ wpack_b,
    const float* __restrict__ bias_f, const float* __restrict__ bias_b,
    _Float16* __restrict__ h_f, _Float16* __restrict__ h_b, int L) {
    __shared__ __align__(16) _Float16 a_buf[16 * AS];

    int dir = blockIdx.y;
    int bt  = blockIdx.x;
    const _Float16* W    = dir ? wpack_b : wpack_f;
    const float*    bias = dir ? bias_b  : bias_f;
    _Float16*       hout = dir ? h_b     : h_f;

    int tid  = threadIdx.x;
    int lane = tid & 31;
    int wave = tid >> 5;          // == hidden tile ht (0..15)
    int mlo  = lane & 15;
    int kb   = (lane & 16) ? 8 : 0;
    int mbase = (lane & 16) ? 8 : 0;

    // h region := 0
    for (int e = tid; e < 16 * HID; e += 512) {
        a_buf[(e >> 8) * AS + (e & 255)] = (_Float16)0.0f;
    }

    // per-thread x staging slot: row xr (0..15), 4 f16 at column xc4
    int xr  = tid >> 5;
    int xc4 = (tid & 31) * 4;
    const _Float16* xsrc_base = exp16 + ((size_t)(bt * 16 + xr) * L) * EMB + xc4;

    // preload x for step 0 into registers
    v4h xv = *(const v4h*)(xsrc_base + (size_t)(dir ? (L - 1) : 0) * EMB);

    float c_state[8];
#pragma unroll
    for (int r = 0; r < 8; ++r) c_state[r] = 0.0f;

    int j  = wave * 16 + mlo;      // hidden unit owned by this thread-column
    float bi = bias[j],           bf = bias[HID + j];
    float bg = bias[2 * HID + j], bo = bias[3 * HID + j];

    for (int step = 0; step < L; ++step) {
        int l = dir ? (L - 1 - step) : step;

        // stage this step's x into LDS (safe: previous step's readers passed barrier 2)
        *(v4h*)(a_buf + xr * AS + HID + xc4) = xv;
        // issue next step's x load now; consumed at next loop top
        if (step + 1 < L) {
            int ln = dir ? (L - 2 - step) : (step + 1);
            xv = *(const v4h*)(xsrc_base + (size_t)ln * EMB);
        }

        __syncthreads();                          // h + x staged for this step

        v8f acc[4];
#pragma unroll
        for (int g = 0; g < 4; ++g) acc[g] = v8f_zero();

#pragma unroll 2
        for (int kt = 0; kt < 12; ++kt) {
            const v8h* ap = (const v8h*)(a_buf + mlo * AS + kt * 32 + kb);
            v16h afrag = combine16(ap[0], ap[2]);
#pragma unroll
            for (int g = 0; g < 4; ++g) {
                int nt = g * 16 + wave;
                const _Float16* tb = W + ((size_t)(nt * 12 + kt)) * 512 + lane * 16;
                v16h bfrag = combine16(*(const v8h*)tb, *(const v8h*)(tb + 8));
                acc[g] = wmma_f16(afrag, bfrag, acc[g]);
            }
        }

        __syncthreads();                          // all waves done reading a_buf

        // gate nonlinearity, cell update, write h back (LDS A-layout + global f16)
#pragma unroll
        for (int r = 0; r < 8; ++r) {
            float ig = fast_sigmoid(acc[0][r] + bi);
            float fg = fast_sigmoid(acc[1][r] + bf);
            float gg = fast_tanh(acc[2][r] + bg);
            float og = fast_sigmoid(acc[3][r] + bo);
            float c  = fg * c_state[r] + ig * gg;
            c_state[r] = c;
            _Float16 h16 = (_Float16)(og * fast_tanh(c));
            int m = mbase + r;
            a_buf[m * AS + j] = h16;
            hout[((size_t)((bt * 16 + m) * L + l)) * HID + j] = h16;
        }
    }
}

// ---------------- final projection: [h_f|h_b] (M x 512) @ lin_w^T -> (M x 80) ----------------
// block = 160 threads (5 waves, one per 16-col tile of MEL); grid = M/16 = 4*L.

__global__ __launch_bounds__(160, 1) void k_final(
    const _Float16* __restrict__ h_f, const _Float16* __restrict__ h_b,
    const _Float16* __restrict__ linpack, const float* __restrict__ lin_b,
    float* __restrict__ out, int L) {
    int lane = threadIdx.x & 31;
    int nt   = threadIdx.x >> 5;   // 0..4
    int rt   = blockIdx.x;
    int mlo  = lane & 15;
    int kb   = (lane & 16) ? 8 : 0;
    int r_glob = rt * 16 + mlo;    // flattened (b*L + l) row index for A loads

    const _Float16* arow_f = h_f + (size_t)r_glob * HID;
    const _Float16* arow_b = h_b + (size_t)r_glob * HID;

    v8f acc = v8f_zero();
#pragma unroll 2
    for (int kt = 0; kt < 16; ++kt) {
        const _Float16* arow = (kt < 8) ? (arow_f + kt * 32) : (arow_b + (kt - 8) * 32);
        v16h afrag = combine16(*(const v8h*)(arow + kb), *(const v8h*)(arow + kb + 16));
        const _Float16* tb = linpack + ((size_t)(nt * 16 + kt)) * 512 + lane * 16;
        v16h bfrag = combine16(*(const v8h*)tb, *(const v8h*)(tb + 8));
        acc = wmma_f16(afrag, bfrag, acc);
    }
    int n = nt * 16 + mlo;
    float bn = lin_b[n];
    int mb = (lane & 16) ? 8 : 0;
#pragma unroll
    for (int r = 0; r < 8; ++r) {
        int mg = rt * 16 + mb + r;
        out[(size_t)mg * MEL + n] = acc[r] + bn;
    }
}

// ---------------- launch ----------------

extern "C" void kernel_launch(void* const* d_in, const int* in_sizes, int n_in,
                              void* d_out, int out_size, void* d_ws, size_t ws_size,
                              hipStream_t stream) {
    (void)in_sizes; (void)n_in; (void)ws_size;
    const int*   x     = (const int*)d_in[0];
    const float* embed = (const float*)d_in[1];
    const float* dp_w  = (const float*)d_in[2];
    const float* dp_b  = (const float*)d_in[3];
    const float* wih_f = (const float*)d_in[4];
    const float* whh_f = (const float*)d_in[5];
    const float* bih_f = (const float*)d_in[6];
    const float* bhh_f = (const float*)d_in[7];
    const float* wih_b = (const float*)d_in[8];
    const float* whh_b = (const float*)d_in[9];
    const float* bih_b = (const float*)d_in[10];
    const float* bhh_b = (const float*)d_in[11];
    const float* lin_w = (const float*)d_in[12];
    const float* lin_b = (const float*)d_in[13];
    float*       out   = (float*)d_out;

    int L = out_size / (BB * MEL);
    if (L < 1) L = 1;

    char* ws = (char*)d_ws;
    size_t off = 0;
    auto alloc = [&](size_t bytes) -> char* {
        char* p = ws + off;
        off = (off + bytes + 255) & ~(size_t)255;
        return p;
    };
    int*       cum     = (int*)alloc((size_t)BB * TT * 4);
    _Float16*  embed16 = (_Float16*)alloc((size_t)VOCAB * EMB * 2);
    float*     bias_f  = (float*)alloc((size_t)GH * 4);
    float*     bias_b  = (float*)alloc((size_t)GH * 4);
    _Float16*  wpack_f = (_Float16*)alloc((size_t)64 * 12 * 512 * 2);
    _Float16*  wpack_b = (_Float16*)alloc((size_t)64 * 12 * 512 * 2);
    _Float16*  linpack = (_Float16*)alloc((size_t)5 * 16 * 512 * 2);
    _Float16*  exp16   = (_Float16*)alloc((size_t)BB * L * EMB * 2);
    _Float16*  h_f     = (_Float16*)alloc((size_t)BB * L * HID * 2);
    _Float16*  h_b     = (_Float16*)alloc((size_t)BB * L * HID * 2);

    k_prep_embed<<<(VOCAB * EMB + 255) / 256, 256, 0, stream>>>(embed, embed16);
    k_prep_bias<<<(GH + 255) / 256, 256, 0, stream>>>(bih_f, bhh_f, bias_f);
    k_prep_bias<<<(GH + 255) / 256, 256, 0, stream>>>(bih_b, bhh_b, bias_b);
    k_pack_gates<<<(64 * 12 * 512 + 255) / 256, 256, 0, stream>>>(whh_f, wih_f, wpack_f);
    k_pack_gates<<<(64 * 12 * 512 + 255) / 256, 256, 0, stream>>>(whh_b, wih_b, wpack_b);
    k_pack_lin<<<(5 * 16 * 512 + 255) / 256, 256, 0, stream>>>(lin_w, linpack);
    k_dur_cum<<<BB, TT, 0, stream>>>(x, embed, dp_w, dp_b, cum);
    k_expand<<<dim3((L + 7) / 8, BB), 256, 0, stream>>>(x, cum, embed16, exp16, L);
    k_lstm<<<dim3(4, 2), 512, 0, stream>>>(exp16, wpack_f, wpack_b, bias_f, bias_b,
                                           h_f, h_b, L);
    k_final<<<4 * L, 160, 0, stream>>>(h_f, h_b, linpack, lin_b, out, L);
}
